// LM_68710886801530
// MI455X (gfx1250) — compile-verified, matrix operand order checked by reading, not run on hardware
//
#include <hip/hip_runtime.h>
#include <math.h>

// ---------------- model constants ----------------
#define E_    1024
#define H_    8
#define HS_   128
#define L_    8
#define V_    30000
#define VP_   30080           // V padded to multiple of 128 (block tile N)
#define B_    8
#define T_    512
#define BT_   4096            // B*T
#define FF_   4096            // 4*E
#define EPS_  1e-5f
#define LDAP  40              // LDS row pitch (elems): 32 data + 8 pad (80B)

typedef __bf16 bf16;
typedef __attribute__((ext_vector_type(8)))  __bf16 v8bf;
typedef __attribute__((ext_vector_type(16))) __bf16 v16bf;
typedef __attribute__((ext_vector_type(8)))  float  v8f;

// ============================================================
// Async global->LDS 16B copy (gfx1250, ASYNCcnt-tracked)
// ============================================================
__device__ __forceinline__ void async_cp16(unsigned lds_addr, const bf16* g) {
  asm volatile("global_load_async_to_lds_b128 %0, %1, off"
               :: "v"(lds_addr), "v"(g) : "memory");
}
__device__ __forceinline__ void wait_async_le4() {
  asm volatile("s_wait_asynccnt 0x4" ::: "memory");
}
__device__ __forceinline__ void wait_async_0() {
  asm volatile("s_wait_asynccnt 0x0" ::: "memory");
}
__device__ __forceinline__ unsigned lds_off(const bf16* p) {
  return (unsigned)(size_t)p;   // LDS aperture: low 32 bits = LDS address
}

// 16-bit WMMA fragment from LDS tile (row pitch LDAP):
// lanes 0-15 : K = 0..7 , 16..23 ; lanes 16-31: K = 8..15, 24..31
__device__ __forceinline__ v16bf lds_frag(const bf16* p) {
  v8bf lo = *(const v8bf*)(p);
  v8bf hi = *(const v8bf*)(p + 16);
  return __builtin_shufflevector(lo, hi, 0,1,2,3,4,5,6,7,8,9,10,11,12,13,14,15);
}

// ============================================================
// Block-cooperative bf16 WMMA GEMM, LDS double-buffered with
// async copies.  C[m,n] = sum_k A[m,k]*Bt[n,k] (+bias)(relu)(+resid)
// Block: 256 thr = 8 waves -> 128x128 C tile; wave: 32x64.
// Requires: M %128==0, Npad %128==0, K %32==0 (stores guarded by Nstore).
// ============================================================
__global__ void __launch_bounds__(256)
gemm_bf16_wmma(
    const bf16* __restrict__ A, const bf16* __restrict__ Bt,
    const float* __restrict__ bias, const float* __restrict__ resid,
    float* __restrict__ Cf, bf16* __restrict__ Cb,
    int M, int Npad, int K, int Nstore,
    int lda, int ldb, int ldc,
    long long sAb, long long sAh, long long sBb, long long sBh,
    long long sCb, long long sCh,
    int Hdim, int relu)
{
  __shared__ bf16 sA[2][128 * LDAP];
  __shared__ bf16 sB[2][128 * LDAP];

  const int z  = blockIdx.y;
  const int bb = z / Hdim, hh = z % Hdim;
  const bf16* Ab = A  + (long long)bb * sAb + (long long)hh * sAh;
  const bf16* Bb = Bt + (long long)bb * sBb + (long long)hh * sBh;
  const long long coff = (long long)bb * sCb + (long long)hh * sCh;

  const int tid  = threadIdx.x;
  const int lane = tid & 31;
  const int wave = tid >> 5;               // 0..7
  const int wm   = wave >> 1;              // 0..3 -> 32-row band
  const int wn   = wave & 1;               // 0..1 -> 64-col band
  const int half = lane >> 4;
  const int r16  = lane & 15;

  const int Mt = M >> 7;
  const int tM = (blockIdx.x % Mt) << 7;
  const int tN = (blockIdx.x / Mt) << 7;

  // cooperative stage of one 128x32 A tile + 128x32 B tile (4 async/thread)
  const int srow = tid >> 2;               // 0..63
  const int scol = (tid & 3) * 8;          // 0,8,16,24
  auto stage = [&](int buf, int k0) {
#pragma unroll
    for (int i = 0; i < 2; ++i) {
      const int row = srow + i * 64;
      async_cp16(lds_off(&sA[buf][row * LDAP + scol]),
                 Ab + (size_t)(tM + row) * lda + k0 + scol);
      async_cp16(lds_off(&sB[buf][row * LDAP + scol]),
                 Bb + (size_t)(tN + row) * ldb + k0 + scol);
    }
  };

  v8f acc[2][4] = {};

  int buf = 0;
  stage(0, 0);
  for (int k0 = 0; k0 < K; k0 += 32) {
    if (k0 + 32 < K) { stage(buf ^ 1, k0 + 32); wait_async_le4(); }
    else             { wait_async_0(); }
    __syncthreads();

    // preload ALL fragments for this K-step, then issue 8 WMMAs back-to-back
    v16bf af[2], bfrag[4];
#pragma unroll
    for (int a = 0; a < 2; ++a)
      af[a] = lds_frag(&sA[buf][(wm * 32 + a * 16 + r16) * LDAP + half * 8]);
#pragma unroll
    for (int g = 0; g < 4; ++g)
      bfrag[g] = lds_frag(&sB[buf][(wn * 64 + g * 16 + r16) * LDAP + half * 8]);
#pragma unroll
    for (int g = 0; g < 4; ++g)
      acc[0][g] = __builtin_amdgcn_wmma_f32_16x16x32_bf16(
          false, af[0], false, bfrag[g], (short)0, acc[0][g], false, false);
#pragma unroll
    for (int g = 0; g < 4; ++g)
      acc[1][g] = __builtin_amdgcn_wmma_f32_16x16x32_bf16(
          false, af[1], false, bfrag[g], (short)0, acc[1][g], false, false);

    __syncthreads();
    buf ^= 1;
  }

  // C/D layout: VGPR r -> lanes 0-15: M=r, lanes 16-31: M=8+r; N = lane&15
#pragma unroll
  for (int a = 0; a < 2; ++a) {
    const int mbase = tM + wm * 32 + a * 16 + half * 8;
#pragma unroll
    for (int g = 0; g < 4; ++g) {
      const int n = tN + wn * 64 + g * 16 + r16;
      if (n >= Nstore) continue;
      const float bv = bias ? bias[n] : 0.0f;
#pragma unroll
      for (int r = 0; r < 8; ++r) {
        const long long ci = coff + (long long)(mbase + r) * ldc + n;
        float v = acc[a][g][r] + bv;
        if (relu) v = v > 0.0f ? v : 0.0f;
        if (resid) v += resid[ci];
        if (Cf) Cf[ci] = v;
        if (Cb) Cb[ci] = (bf16)v;
      }
    }
  }
}

// ============================================================
// Embedding: h[b,t,:] = tok_emb[x[b,t],:] + pos_emb[t,:]
// ============================================================
__global__ void embed_kernel(const int* __restrict__ x,
                             const float* __restrict__ tok,
                             const float* __restrict__ pos,
                             float* __restrict__ h) {
  const int row = blockIdx.x;
  const int t   = row % T_;
  const int id  = x[row];
  const float* tp = tok + (size_t)id * E_;
  const float* pp = pos + (size_t)t * E_;
  float* hp = h + (size_t)row * E_;
  for (int j = threadIdx.x; j < E_; j += 256) hp[j] = tp[j] + pp[j];
}

// ============================================================
// LayerNorm -> bf16 output (block of 256 per row, E=1024)
// ============================================================
__global__ void ln_kernel(const float* __restrict__ x,
                          const float* __restrict__ w,
                          const float* __restrict__ b,
                          bf16* __restrict__ out) {
  __shared__ float sred[256];
  const int row = blockIdx.x;
  const float* xr = x + (size_t)row * E_;
  float v[4];
  float s = 0.0f;
#pragma unroll
  for (int j = 0; j < 4; ++j) { v[j] = xr[threadIdx.x + j * 256]; s += v[j]; }
  sred[threadIdx.x] = s; __syncthreads();
  for (int st = 128; st > 0; st >>= 1) {
    if (threadIdx.x < st) sred[threadIdx.x] += sred[threadIdx.x + st];
    __syncthreads();
  }
  const float mean = sred[0] * (1.0f / E_);
  __syncthreads();
  float vs = 0.0f;
#pragma unroll
  for (int j = 0; j < 4; ++j) { float d = v[j] - mean; vs += d * d; }
  sred[threadIdx.x] = vs; __syncthreads();
  for (int st = 128; st > 0; st >>= 1) {
    if (threadIdx.x < st) sred[threadIdx.x] += sred[threadIdx.x + st];
    __syncthreads();
  }
  const float inv = rsqrtf(sred[0] * (1.0f / E_) + EPS_);
  bf16* op = out + (size_t)row * E_;
#pragma unroll
  for (int j = 0; j < 4; ++j) {
    const int i = threadIdx.x + j * 256;
    op[i] = (bf16)((v[j] - mean) * inv * w[i] + b[i]);
  }
}

// ============================================================
// Convert Wq/Wk/Wv (H,E,HS) f32 -> bf16 [N=E rows, K=E] (n=h*HS+d)
// ============================================================
__global__ void convq_kernel(const float* __restrict__ w, bf16* __restrict__ out) {
  const int idx = blockIdx.x * 256 + threadIdx.x;
  if (idx >= E_ * E_) return;
  const int k = idx & (E_ - 1);
  const int n = idx >> 10;
  out[idx] = (bf16)w[(size_t)(n >> 7) * (E_ * HS_) + (size_t)k * HS_ + (n & 127)];
}

// ============================================================
// Transpose-convert W [K,N] f32 -> bf16 [Npad rows, K] (zero pad)
// ============================================================
__global__ void convt_kernel(const float* __restrict__ w, bf16* __restrict__ out,
                             int K, int N, int Npad) {
  const size_t idx = (size_t)blockIdx.x * 256 + threadIdx.x;
  if (idx >= (size_t)Npad * K) return;
  const int k = (int)(idx % K);
  const int n = (int)(idx / K);
  out[idx] = (n < N) ? (bf16)w[(size_t)k * N + n] : (bf16)0.0f;
}

// ============================================================
// V transpose: v[BT,E] bf16 -> vT[(b*H+h)*HS+d, T] bf16
// ============================================================
__global__ void vtrans_kernel(const bf16* __restrict__ v, bf16* __restrict__ vT) {
  const size_t idx = (size_t)blockIdx.x * 256 + threadIdx.x;
  const int s  = (int)(idx % T_);
  size_t r = idx / T_;
  const int d  = (int)(r % HS_);
  r /= HS_;
  const int h  = (int)(r % H_);
  const int b  = (int)(r / H_);
  vT[idx] = v[((size_t)(b * T_ + s)) * E_ + h * HS_ + d];
}

// ============================================================
// Masked causal softmax row: S[b,h,t,:] f32 -> P bf16
// ============================================================
__global__ void softmax_kernel(const float* __restrict__ S,
                               const float* __restrict__ mask,
                               bf16* __restrict__ P) {
  __shared__ float sred[256];
  const int r = blockIdx.x;
  const int b = r / (H_ * T_);
  const int t = r % T_;
  const float* sr = S + (size_t)r * T_;
  bf16* pr = P + (size_t)r * T_;
  const float scale = 0.08838834764831845f; // HS^-0.5
  float v[2]; bool keep[2];
  float mx = -3.0e38f;
#pragma unroll
  for (int j = 0; j < 2; ++j) {
    const int s = threadIdx.x + j * 256;
    keep[j] = (s <= t) && (mask[b * T_ + s] != 0.0f);
    v[j] = keep[j] ? sr[s] * scale : -1.0e30f;
    mx = fmaxf(mx, v[j]);
  }
  sred[threadIdx.x] = mx; __syncthreads();
  for (int st = 128; st > 0; st >>= 1) {
    if (threadIdx.x < st) sred[threadIdx.x] = fmaxf(sred[threadIdx.x], sred[threadIdx.x + st]);
    __syncthreads();
  }
  mx = sred[0]; __syncthreads();
  float e[2]; float sum = 0.0f;
#pragma unroll
  for (int j = 0; j < 2; ++j) {
    e[j] = keep[j] ? __expf(v[j] - mx) : 0.0f;
    sum += e[j];
  }
  sred[threadIdx.x] = sum; __syncthreads();
  for (int st = 128; st > 0; st >>= 1) {
    if (threadIdx.x < st) sred[threadIdx.x] += sred[threadIdx.x + st];
    __syncthreads();
  }
  const float inv = 1.0f / sred[0];
#pragma unroll
  for (int j = 0; j < 2; ++j) pr[threadIdx.x + j * 256] = (bf16)(e[j] * inv);
}

// ============================================================
// Per-row NLL
// ============================================================
__global__ void nll_kernel(const float* __restrict__ logits,
                           const int* __restrict__ tgt,
                           const float* __restrict__ mask,
                           float* __restrict__ nll) {
  __shared__ float sred[256];
  const int row = blockIdx.x;
  const float* lr = logits + (size_t)row * V_;
  float mx = -3.0e38f;
  for (int i = threadIdx.x; i < V_; i += 256) mx = fmaxf(mx, lr[i]);
  sred[threadIdx.x] = mx; __syncthreads();
  for (int st = 128; st > 0; st >>= 1) {
    if (threadIdx.x < st) sred[threadIdx.x] = fmaxf(sred[threadIdx.x], sred[threadIdx.x + st]);
    __syncthreads();
  }
  mx = sred[0]; __syncthreads();
  float sum = 0.0f;
  for (int i = threadIdx.x; i < V_; i += 256) sum += __expf(lr[i] - mx);
  sred[threadIdx.x] = sum; __syncthreads();
  for (int st = 128; st > 0; st >>= 1) {
    if (threadIdx.x < st) sred[threadIdx.x] += sred[threadIdx.x + st];
    __syncthreads();
  }
  if (threadIdx.x == 0) {
    const float logZ = mx + __logf(sred[0]);
    nll[row] = -(lr[tgt[row]] - logZ) * mask[row];
  }
}

__global__ void loss_kernel(const float* __restrict__ nll,
                            const float* __restrict__ mask,
                            float* __restrict__ out) {
  __shared__ float sa[256], sb[256];
  float a = 0.0f, b = 0.0f;
  for (int i = threadIdx.x; i < BT_; i += 256) { a += nll[i]; b += mask[i]; }
  sa[threadIdx.x] = a; sb[threadIdx.x] = b; __syncthreads();
  for (int st = 128; st > 0; st >>= 1) {
    if (threadIdx.x < st) { sa[threadIdx.x] += sa[threadIdx.x + st];
                            sb[threadIdx.x] += sb[threadIdx.x + st]; }
    __syncthreads();
  }
  if (threadIdx.x == 0) out[0] = sa[0] / sb[0];
}

// ============================================================
// Host launch
// ============================================================
extern "C" void kernel_launch(void* const* d_in, const int* in_sizes, int n_in,
                              void* d_out, int out_size, void* d_ws, size_t ws_size,
                              hipStream_t stream) {
  (void)in_sizes; (void)n_in; (void)out_size; (void)ws_size;

  const int*   x_       = (const int*)d_in[0];
  const int*   targets  = (const int*)d_in[1];
  const float* amask    = (const float*)d_in[2];
  const float* tok_emb  = (const float*)d_in[3];
  const float* pos_emb  = (const float*)d_in[4];
  const float* Wq       = (const float*)d_in[5];
  const float* Wk       = (const float*)d_in[6];
  const float* Wv       = (const float*)d_in[7];
  const float* Wproj    = (const float*)d_in[8];
  const float* bproj    = (const float*)d_in[9];
  const float* ln1_w    = (const float*)d_in[10];
  const float* ln1_b    = (const float*)d_in[11];
  const float* ln2_w    = (const float*)d_in[12];
  const float* ln2_b    = (const float*)d_in[13];
  const float* W1       = (const float*)d_in[14];
  const float* b1       = (const float*)d_in[15];
  const float* W2       = (const float*)d_in[16];
  const float* b2       = (const float*)d_in[17];
  const float* lnf_w    = (const float*)d_in[18];
  const float* lnf_b    = (const float*)d_in[19];
  const float* Whead    = (const float*)d_in[20];
  const float* bhead    = (const float*)d_in[21];

  float* logits = (float*)d_out;                     // [BT, V]
  float* lossp  = logits + (size_t)BT_ * V_;

  // ---- workspace carve ----
  char* p = (char*)d_ws;
  size_t off = 0;
  auto carve = [&](size_t bytes) {
    void* q = p + off;
    off += (bytes + 255) & ~(size_t)255;
    return q;
  };
  float* h    = (float*)carve((size_t)BT_ * E_ * 4);
  bf16*  xn   = (bf16*)carve((size_t)BT_ * E_ * 2);
  bf16*  q    = (bf16*)carve((size_t)BT_ * E_ * 2);
  bf16*  k    = (bf16*)carve((size_t)BT_ * E_ * 2);
  bf16*  v    = (bf16*)carve((size_t)BT_ * E_ * 2);
  bf16*  vT   = (bf16*)carve((size_t)BT_ * E_ * 2);
  bf16*  att  = (bf16*)carve((size_t)BT_ * E_ * 2);
  bf16*  ff   = (bf16*)carve((size_t)BT_ * FF_ * 2);
  float* S    = (float*)carve((size_t)B_ * H_ * T_ * T_ * 4);
  bf16*  P    = (bf16*)carve((size_t)B_ * H_ * T_ * T_ * 2);
  bf16*  wq_t = (bf16*)carve((size_t)E_ * E_ * 2);
  bf16*  wk_t = (bf16*)carve((size_t)E_ * E_ * 2);
  bf16*  wv_t = (bf16*)carve((size_t)E_ * E_ * 2);
  bf16*  wp_t = (bf16*)carve((size_t)E_ * E_ * 2);
  bf16*  w1_t = (bf16*)carve((size_t)E_ * FF_ * 2);
  bf16*  w2_t = (bf16*)carve((size_t)E_ * FF_ * 2);
  bf16*  wh_t = (bf16*)carve((size_t)VP_ * E_ * 2);
  float* nllb = (float*)carve((size_t)BT_ * 4);

  auto gemm = [&](const bf16* A, const bf16* Bt, const float* bias,
                  const float* resid, float* Cf, bf16* Cb,
                  int M, int Npad, int K, int Nstore,
                  int lda, int ldb, int ldc,
                  long long sAb, long long sAh, long long sBb, long long sBh,
                  long long sCb, long long sCh, int Hdim, int relu, int batch) {
    dim3 g((M >> 7) * (Npad >> 7), batch);
    gemm_bf16_wmma<<<g, 256, 0, stream>>>(A, Bt, bias, resid, Cf, Cb,
        M, Npad, K, Nstore, lda, ldb, ldc,
        sAb, sAh, sBb, sBh, sCb, sCh, Hdim, relu);
  };

  embed_kernel<<<BT_, 256, 0, stream>>>(x_, tok_emb, pos_emb, h);

  const long long sTE  = (long long)T_ * E_;
  const long long sTT  = (long long)T_ * T_;
  const long long sHTT = (long long)H_ * T_ * T_;
  const long long sHST = (long long)HS_ * T_;

  for (int l = 0; l < L_; ++l) {
    const float* Wq_l = Wq + (size_t)l * H_ * E_ * HS_;
    const float* Wk_l = Wk + (size_t)l * H_ * E_ * HS_;
    const float* Wv_l = Wv + (size_t)l * H_ * E_ * HS_;

    ln_kernel<<<BT_, 256, 0, stream>>>(h, ln1_w + l * E_, ln1_b + l * E_, xn);

    convq_kernel<<<(E_ * E_) / 256, 256, 0, stream>>>(Wq_l, wq_t);
    convq_kernel<<<(E_ * E_) / 256, 256, 0, stream>>>(Wk_l, wk_t);
    convq_kernel<<<(E_ * E_) / 256, 256, 0, stream>>>(Wv_l, wv_t);

    gemm(xn, wq_t, nullptr, nullptr, nullptr, q, BT_, E_, E_, E_, E_, E_, E_,
         0,0,0,0,0,0, 1, 0, 1);
    gemm(xn, wk_t, nullptr, nullptr, nullptr, k, BT_, E_, E_, E_, E_, E_, E_,
         0,0,0,0,0,0, 1, 0, 1);
    gemm(xn, wv_t, nullptr, nullptr, nullptr, v, BT_, E_, E_, E_, E_, E_, E_,
         0,0,0,0,0,0, 1, 0, 1);

    vtrans_kernel<<<(B_ * H_ * HS_ * T_) / 256, 256, 0, stream>>>(v, vT);

    // S[b,h,t,s] = q . k  (batched over B*H, K=HS)
    gemm(q, k, nullptr, nullptr, S, nullptr, T_, T_, HS_, T_, E_, E_, T_,
         sTE, HS_, sTE, HS_, sHTT, sTT, H_, 0, B_ * H_);

    softmax_kernel<<<B_ * H_ * T_, 256, 0, stream>>>(S, amask, P);

    // att[b,t,h*HS+d] = P @ V  (batched, K=T)
    gemm(P, vT, nullptr, nullptr, nullptr, att, T_, HS_, T_, HS_, T_, T_, E_,
         sHTT, sTT, sHST, sHST, sTE, (long long)HS_, H_, 0, B_ * H_);

    convt_kernel<<<(E_ * E_) / 256, 256, 0, stream>>>(
        Wproj + (size_t)l * E_ * E_, wp_t, E_, E_, E_);
    gemm(att, wp_t, bproj + l * E_, h, h, nullptr, BT_, E_, E_, E_, E_, E_, E_,
         0,0,0,0,0,0, 1, 0, 1);

    ln_kernel<<<BT_, 256, 0, stream>>>(h, ln2_w + l * E_, ln2_b + l * E_, xn);

    convt_kernel<<<((E_ * FF_) / 256), 256, 0, stream>>>(
        W1 + (size_t)l * E_ * FF_, w1_t, E_, FF_, FF_);
    gemm(xn, w1_t, b1 + l * FF_, nullptr, nullptr, ff, BT_, FF_, E_, FF_,
         E_, E_, FF_, 0,0,0,0,0,0, 1, 1, 1);

    convt_kernel<<<((FF_ * E_) / 256), 256, 0, stream>>>(
        W2 + (size_t)l * FF_ * E_, w2_t, FF_, E_, E_);
    gemm(ff, w2_t, b2 + l * E_, h, h, nullptr, BT_, E_, FF_, E_,
         FF_, FF_, E_, 0,0,0,0,0,0, 1, 0, 1);
  }

  ln_kernel<<<BT_, 256, 0, stream>>>(h, lnf_w, lnf_b, xn);

  {
    const size_t tot = (size_t)VP_ * E_;
    convt_kernel<<<(int)((tot + 255) / 256), 256, 0, stream>>>(Whead, wh_t, E_, V_, VP_);
  }
  gemm(xn, wh_t, bhead, nullptr, logits, nullptr, BT_, VP_, E_, V_,
       E_, E_, V_, 0,0,0,0,0,0, 1, 0, 1);

  nll_kernel<<<BT_, 256, 0, stream>>>(logits, targets, amask, nllb);
  loss_kernel<<<1, 256, 0, stream>>>(nllb, amask, lossp);
}
